// GPT_16681652978393
// MI455X (gfx1250) — matrix operand order repaired and checked by reading, not compiled, not run on hardware
//
#include <hip/hip_runtime.h>

// GPT-2 small forward on gfx1250 (MI455X): bf16 WMMA, fp32 accum.
// Memory-bound (~290 GFLOP vs ~500MB fp32 params @ 23.3TB/s): weights are
// converted to bf16 N-major once per call; GEMMs stream bf16 through
// v_wmma_f32_16x16x32_bf16 with double-buffered LDS tiles. BOTH panels are
// staged by the Tensor Data Mover (tensor_load_to_lds, one descriptor per
// issuing wave, s_wait_tensorcnt) fully overlapped with WMMA compute; the
// GEMM inner loop is pure LDS-b128 + WMMA.

#define SV   50257
#define DM   768
#define NH   12
#define NL   12
#define SQ   1024
#define FF   3072
#define DHD  64
#define EPSLN 1e-5f

typedef __bf16 bf16;
typedef __attribute__((ext_vector_type(16))) __bf16 v16bf;
typedef __attribute__((ext_vector_type(8)))  __bf16 bf16x8;
typedef __attribute__((ext_vector_type(8)))  float  v8f;

__device__ __forceinline__ v8f vzero8() {
  v8f z; for (int i = 0; i < 8; ++i) z[i] = 0.f; return z;
}
__device__ __forceinline__ v16bf make_frag(bf16x8 lo, bf16x8 hi) {
  v16bf a;
  for (int i = 0; i < 8; ++i) { a[i] = lo[i]; a[i + 8] = hi[i]; }
  return a;
}
// Force a (wave-uniform) value into an SGPR-resident def for "s" asm operands;
// inline asm does NOT auto-insert readfirstlane for divergent-looking values.
__device__ __forceinline__ unsigned sgpr(unsigned x) {
  return (unsigned)__builtin_amdgcn_readfirstlane((int)x);
}

// --------------------------------------------------------------------- TDM
// 2D tile load Global->LDS via Tensor Data Mover. data_size=2B. LDS rows are
// padded 32->40 bf16 via D# padding: +4 DWORDs after every 16 DWORDs
// (pad_interval=3 -> 16 DW, pad_amount=3 -> 4 DW). Rows beyond tensorRows
// (OOB vs tensor_dim1) are zero-filled by the TDM. Issue once per wave.
// All descriptor words are readfirstlane'd into SGPRs, then assembled into
// pinned groups s[80:83] / s[84:91] for the wide-tuple operands.
__device__ __forceinline__ void tdm_load_2d(unsigned lds_off, const void* gp,
                                            unsigned tileK, unsigned tileRows,
                                            unsigned tensorRows,
                                            unsigned long rowStrideElems) {
  unsigned long ga = (unsigned long)gp;
  unsigned g00 = sgpr(1u);                             // count=1
  unsigned g01 = sgpr(lds_off);                        // lds_addr (bytes)
  unsigned g02 = sgpr((unsigned)ga);                   // global_addr[31:0]
  unsigned g03 = sgpr((unsigned)((ga >> 32) & 0x01FFFFFFu)  // addr[56:32]
                      | (2u << 30));                   // type=2 ("image")
  unsigned g10 = sgpr((1u << 16)                       // data_size = 2B
                      | (1u << 20)                     // pad_enable
                      | (3u << 22)                     // pad_interval: 16 DW
                      | (3u << 25));                   // pad_amount : 4 DW
  unsigned g11 = sgpr((tileK & 0xFFFFu) << 16);        // tensor_dim0[15:0]
  unsigned g12 = sgpr((tileK >> 16) | ((tensorRows & 0xFFFFu) << 16));
  unsigned g13 = sgpr((tensorRows >> 16) | (tileK << 16)); // dim1 hi|tile_dim0
  unsigned g14 = sgpr(tileRows);                       // tile_dim1, tile_dim2=0
  unsigned g15 = sgpr((unsigned)rowStrideElems);       // dim0_stride lo32
  unsigned g16 = sgpr((unsigned)(rowStrideElems >> 32) & 0xFFFFu);
  asm volatile(
      "s_mov_b32 s80, %0\n\t"
      "s_mov_b32 s81, %1\n\t"
      "s_mov_b32 s82, %2\n\t"
      "s_mov_b32 s83, %3\n\t"
      "s_mov_b32 s84, %4\n\t"
      "s_mov_b32 s85, %5\n\t"
      "s_mov_b32 s86, %6\n\t"
      "s_mov_b32 s87, %7\n\t"
      "s_mov_b32 s88, %8\n\t"
      "s_mov_b32 s89, %9\n\t"
      "s_mov_b32 s90, %10\n\t"
      "s_mov_b32 s91, 0\n\t"
      "tensor_load_to_lds s[80:83], s[84:91]"
      :: "s"(g00), "s"(g01), "s"(g02), "s"(g03),
         "s"(g10), "s"(g11), "s"(g12), "s"(g13),
         "s"(g14), "s"(g15), "s"(g16)
      : "memory", "s80", "s81", "s82", "s83", "s84", "s85",
        "s86", "s87", "s88", "s89", "s90", "s91");
}

// ---------------------------------------------------------------- converters
__global__ __launch_bounds__(256)
void cvt_kernel(const float* __restrict__ src, bf16* __restrict__ dst, long n) {
  long i = (long)blockIdx.x * blockDim.x + threadIdx.x;
  if (i < n) dst[i] = (bf16)src[i];
}

// src: [K][N] fp32 (row-major weight), dst: [N][K] bf16 (N-major for WMMA B)
__global__ __launch_bounds__(256)
void transpose_cvt_kernel(const float* __restrict__ src, bf16* __restrict__ dst,
                          int K, int N, long srcStride, long dstStride) {
  __shared__ float tile[32][33];
  const float* s = src + blockIdx.z * srcStride;
  bf16*        d = dst + blockIdx.z * dstStride;
  int nb = blockIdx.x * 32, kb = blockIdx.y * 32;
  int tx = threadIdx.x, ty = threadIdx.y;
  for (int i = ty; i < 32; i += 8)
    tile[i][tx] = s[(long)(kb + i) * N + nb + tx];
  __syncthreads();
  for (int i = ty; i < 32; i += 8)
    d[(long)(nb + i) * K + kb + tx] = (bf16)tile[tx][i];
}

// ------------------------------------------------------------------- embed
__global__ __launch_bounds__(256)
void embed_kernel(const int* __restrict__ tok, const float* __restrict__ wte,
                  const float* __restrict__ wpe, float* __restrict__ x) {
  int s = blockIdx.x;
  int t = tok[s];
  for (int d = threadIdx.x; d < DM; d += blockDim.x)
    x[(long)s * DM + d] = wte[(long)t * DM + d] + wpe[(long)s * DM + d];
}

// ---------------------------------------------------------------- layernorm
__global__ __launch_bounds__(256)
void layernorm_kernel(const float* __restrict__ x, const float* __restrict__ g,
                      const float* __restrict__ b, bf16* __restrict__ out) {
  __shared__ float s1[256], s2[256];
  int s = blockIdx.x;
  const float* xr = x + (long)s * DM;
  float v0[3], sum = 0.f, ssq = 0.f;
  for (int i = 0; i < 3; ++i) {
    float v = xr[threadIdx.x + i * 256];
    v0[i] = v; sum += v; ssq += v * v;
  }
  s1[threadIdx.x] = sum; s2[threadIdx.x] = ssq;
  __syncthreads();
  for (int st = 128; st > 0; st >>= 1) {
    if (threadIdx.x < st) {
      s1[threadIdx.x] += s1[threadIdx.x + st];
      s2[threadIdx.x] += s2[threadIdx.x + st];
    }
    __syncthreads();
  }
  float mu   = s1[0] / DM;
  float var  = s2[0] / DM - mu * mu;
  float rstd = rsqrtf(var + EPSLN);
  for (int i = 0; i < 3; ++i) {
    int d = threadIdx.x + i * 256;
    out[(long)s * DM + d] = (bf16)((v0[i] - mu) * rstd * g[d] + b[d]);
  }
}

// -------------------------------------------------------------------- GEMM
// C[M,N] = A[M,K](bf16,row-major) x B[N,K](bf16,N-major) + bias. Epilogues:
//   MODE 0: outF = acc (+bias)                       (logits)
//   MODE 1: outF = resid + acc + bias                (residual update)
//   MODE 2: outB = bf16(gelu_tanh(acc + bias))       (MLP up-proj)
//   MODE 3: bf16 QKV split -> q/k [H][S][DH], v [H][DH][S]
// 256 thr = 8 waves (wave32) in 2x4 grid; block tile 128x128, wave 64x32,
// BK=32; LDS double-buffered; A panel DMA'd by wave 0's TDM descriptor,
// B panel by wave 1's (tensor_dim1 gives zero-fill past N for logits).
template <int MODE>
__global__ __launch_bounds__(256)
void gemm_bf16_kernel(const bf16* __restrict__ A, const bf16* __restrict__ B,
                      const float* __restrict__ bias, const float* resid,
                      float* outF, bf16* outB, bf16* qbp, bf16* kbp, bf16* vtp,
                      int M, int N, int K) {
  __shared__ __attribute__((aligned(16))) bf16 As[2][128][40];
  __shared__ __attribute__((aligned(16))) bf16 Bs[2][128][40];

  const int t    = threadIdx.x;
  const int lane = t & 31, wid = t >> 5;
  const int r = lane & 15, hi = lane >> 4;
  const int wm = wid & 1, wn = wid >> 1;    // 2x4 wave grid
  const int m0 = blockIdx.y * 128;
  const int n0 = blockIdx.x * 128;
  const unsigned bRows =
      (unsigned)(N - n0 < 128 ? (N - n0 > 0 ? N - n0 : 0) : 128);

  v8f acc[4][2];
  for (int mi = 0; mi < 4; ++mi)
    for (int ni = 0; ni < 2; ++ni) acc[mi][ni] = vzero8();

  auto stage = [&](int buf, int k0) {
    if (wid == 0)
      tdm_load_2d((unsigned)(size_t)&As[buf][0][0], &A[(long)m0 * K + k0],
                  32u, 128u, 128u, (unsigned long)K);
    else if (wid == 1)
      tdm_load_2d((unsigned)(size_t)&Bs[buf][0][0], &B[(long)n0 * K + k0],
                  32u, 128u, bRows, (unsigned long)K);
  };

  stage(0, 0);
  if (wid < 2) __builtin_amdgcn_s_wait_tensorcnt(0);
  __syncthreads();

  int buf = 0;
  for (int k0 = 0; k0 < K; k0 += 32) {
    const bool more = (k0 + 32) < K;
    if (more) stage(buf ^ 1, k0 + 32);

    v16bf af[4], bfg[2];
    for (int mi = 0; mi < 4; ++mi) {  // A layout: k = hi*8+i | 16+hi*8+i
      int ar = wm * 64 + mi * 16 + r;
      bf16x8 lo = *(const bf16x8*)&As[buf][ar][hi * 8];
      bf16x8 hv = *(const bf16x8*)&As[buf][ar][16 + hi * 8];
      af[mi] = make_frag(lo, hv);
    }
    for (int ni = 0; ni < 2; ++ni) {  // B layout: k = hi*8+i | 16+hi*8+i
      int br = wn * 32 + ni * 16 + r;
      bf16x8 lo = *(const bf16x8*)&Bs[buf][br][hi * 8];
      bf16x8 hv = *(const bf16x8*)&Bs[buf][br][16 + hi * 8];
      bfg[ni] = make_frag(lo, hv);
    }
    for (int mi = 0; mi < 4; ++mi)
      for (int ni = 0; ni < 2; ++ni)
        acc[mi][ni] = __builtin_amdgcn_wmma_f32_16x16x32_bf16(
            false, af[mi], false, bfg[ni], (short)0, acc[mi][ni], false, false);

    if (more && wid < 2) __builtin_amdgcn_s_wait_tensorcnt(0);
    __syncthreads();
    buf ^= 1;
  }

  for (int mi = 0; mi < 4; ++mi)
    for (int ni = 0; ni < 2; ++ni) {
      int col = n0 + wn * 32 + ni * 16 + r;
      if (col >= N) continue;
      float bs = bias ? bias[col] : 0.f;
      int rowb = m0 + wm * 64 + mi * 16 + hi * 8;
      if (MODE == 3) {
        int seg = col / DM;
        int d   = col - seg * DM;
        int hh  = d >> 6, dh = d & 63;
        for (int v = 0; v < 8; ++v) {
          float val = acc[mi][ni][v] + bs;
          int row = rowb + v;
          if (seg == 0)      qbp[((long)hh * SQ + row) * DHD + dh] = (bf16)val;
          else if (seg == 1) kbp[((long)hh * SQ + row) * DHD + dh] = (bf16)val;
          else               vtp[((long)hh * DHD + dh) * SQ + row] = (bf16)val;
        }
      } else {
        for (int v = 0; v < 8; ++v) {
          long idx = (long)(rowb + v) * N + col;
          float val = acc[mi][ni][v] + bs;
          if (MODE == 0) {
            outF[idx] = val;
          } else if (MODE == 1) {
            outF[idx] = resid[idx] + val;
          } else {
            float u  = val;
            float th = tanhf(0.7978845608028654f * (u + 0.044715f * u * u * u));
            outB[idx] = (bf16)(0.5f * u * (1.f + th));
          }
        }
      }
    }
}

// --------------------------------------------------------------- attention
// One wave per (head, 16-query tile). Causal flash attention:
//  S = Q Kt (2 WMMAs per 16-key tile, DH=64 = 2 k-steps), online softmax with
//  16-lane butterflies, P relayout C->A via 1KB LDS, O += P V (4 WMMAs/32 keys).
__global__ __launch_bounds__(32)
void attention_kernel(const bf16* __restrict__ qb, const bf16* __restrict__ kb,
                      const bf16* __restrict__ vt, bf16* __restrict__ o) {
  __shared__ __attribute__((aligned(16))) bf16 Pp[16][40];
  const int lane = threadIdx.x & 31;
  const int r = lane & 15, hi = lane >> 4;
  const int h  = blockIdx.y;
  const int q0 = blockIdx.x * 16;

  const bf16* qh = qb + (long)h * SQ * DHD;
  const bf16* kh = kb + (long)h * SQ * DHD;
  const bf16* vh = vt + (long)h * DHD * SQ;

  v16bf aq[2];
  for (int kk = 0; kk < 2; ++kk) {
    const bf16* p = qh + (long)(q0 + r) * DHD + kk * 32;
    bf16x8 lo = *(const bf16x8*)(p + hi * 8);
    bf16x8 hv = *(const bf16x8*)(p + 16 + hi * 8);
    aq[kk] = make_frag(lo, hv);
  }

  float m[8], l[8];
  for (int v = 0; v < 8; ++v) { m[v] = -1e30f; l[v] = 0.f; }
  v8f accv[4];
  for (int nn = 0; nn < 4; ++nn) accv[nn] = vzero8();

  const int nIter = (q0 + 16 + 31) >> 5;
  for (int jb = 0; jb < nIter; ++jb) {
    const int j0 = jb * 32;
    v8f sc[2];
    for (int tt = 0; tt < 2; ++tt) {
      v16bf bk[2];
      for (int kk = 0; kk < 2; ++kk) {
        const bf16* p = kh + (long)(j0 + tt * 16 + r) * DHD + kk * 32;
        bf16x8 lo = *(const bf16x8*)(p + hi * 8);
        bf16x8 hv = *(const bf16x8*)(p + 16 + hi * 8);
        bk[kk] = make_frag(lo, hv);
      }
      v8f s = vzero8();
      s = __builtin_amdgcn_wmma_f32_16x16x32_bf16(false, aq[0], false, bk[0],
                                                  (short)0, s, false, false);
      s = __builtin_amdgcn_wmma_f32_16x16x32_bf16(false, aq[1], false, bk[1],
                                                  (short)0, s, false, false);
      sc[tt] = s;
    }
    float p0[8], p1[8], mnew[8];
    for (int v = 0; v < 8; ++v) {
      int row = q0 + hi * 8 + v;
      int c0 = j0 + r, c1 = j0 + 16 + r;
      float s0 = sc[0][v] * 0.125f; if (c0 > row) s0 = -1e30f;
      float s1 = sc[1][v] * 0.125f; if (c1 > row) s1 = -1e30f;
      float rm = fmaxf(s0, s1);
      for (int off = 8; off >= 1; off >>= 1)
        rm = fmaxf(rm, __shfl_xor(rm, off, 32));
      mnew[v] = fmaxf(m[v], rm);
      p0[v] = __expf(s0 - mnew[v]);
      p1[v] = __expf(s1 - mnew[v]);
    }
    for (int v = 0; v < 8; ++v) {
      float rs = p0[v] + p1[v];
      for (int off = 8; off >= 1; off >>= 1) rs += __shfl_xor(rs, off, 32);
      float sf = __expf(m[v] - mnew[v]);
      l[v] = l[v] * sf + rs;
      m[v] = mnew[v];
      for (int nn = 0; nn < 4; ++nn) accv[nn][v] *= sf;
    }
    for (int v = 0; v < 8; ++v) {             // C-layout -> LDS
      Pp[hi * 8 + v][r]      = (bf16)p0[v];
      Pp[hi * 8 + v][16 + r] = (bf16)p1[v];
    }
    __syncthreads();
    v16bf ap;
    {
      bf16x8 lo = *(const bf16x8*)&Pp[r][hi * 8];
      bf16x8 hv = *(const bf16x8*)&Pp[r][16 + hi * 8];
      ap = make_frag(lo, hv);
    }
    for (int nn = 0; nn < 4; ++nn) {
      const bf16* p = vh + (long)(nn * 16 + r) * SQ + j0;
      bf16x8 lo = *(const bf16x8*)(p + hi * 8);
      bf16x8 hv = *(const bf16x8*)(p + 16 + hi * 8);
      v16bf bv = make_frag(lo, hv);
      accv[nn] = __builtin_amdgcn_wmma_f32_16x16x32_bf16(
          false, ap, false, bv, (short)0, accv[nn], false, false);
    }
    __syncthreads();
  }
  for (int nn = 0; nn < 4; ++nn)
    for (int v = 0; v < 8; ++v) {
      int row = q0 + hi * 8 + v;
      int dh  = nn * 16 + r;
      o[(long)row * DM + h * DHD + dh] = (bf16)(accv[nn][v] / l[v]);
    }
}

// ------------------------------------------------------------------- driver
extern "C" void kernel_launch(void* const* d_in, const int* in_sizes, int n_in,
                              void* d_out, int out_size, void* d_ws,
                              size_t ws_size, hipStream_t stream) {
  const int*   tokens = (const int*)d_in[0];
  const float* wte    = (const float*)d_in[1];
  const float* wpe    = (const float*)d_in[2];
  const float* Wqkv   = (const float*)d_in[3];
  const float* bqkv   = (const float*)d_in[4];
  const float* Wo     = (const float*)d_in[5];
  const float* bo     = (const float*)d_in[6];
  const float* ln1g   = (const float*)d_in[7];
  const float* ln1b   = (const float*)d_in[8];
  const float* ln2g   = (const float*)d_in[9];
  const float* ln2b   = (const float*)d_in[10];
  const float* W1     = (const float*)d_in[11];
  const float* b1     = (const float*)d_in[12];
  const float* W2     = (const float*)d_in[13];
  const float* b2     = (const float*)d_in[14];
  const float* lnfg   = (const float*)d_in[15];
  const float* lnfb   = (const float*)d_in[16];

  char* ws = (char*)d_ws;
  size_t off = 0;
  auto carve = [&](size_t bytes) -> void* {
    void* p = ws + off;
    off = (off + bytes + 255) & ~(size_t)255;
    return p;
  };

  bf16*  wqkvT = (bf16*)carve((size_t)NL * 3 * DM * DM * sizeof(bf16));
  bf16*  woT   = (bf16*)carve((size_t)NL * DM * DM * sizeof(bf16));
  bf16*  w1T   = (bf16*)carve((size_t)NL * FF * DM * sizeof(bf16));
  bf16*  w2T   = (bf16*)carve((size_t)NL * DM * FF * sizeof(bf16));
  bf16*  wteB  = (bf16*)carve((size_t)SV * DM * sizeof(bf16));
  float* x     = (float*)carve((size_t)SQ * DM * sizeof(float));
  bf16*  hbuf  = (bf16*)carve((size_t)SQ * DM * sizeof(bf16));
  bf16*  qb    = (bf16*)carve((size_t)NH * SQ * DHD * sizeof(bf16));
  bf16*  kbuf  = (bf16*)carve((size_t)NH * SQ * DHD * sizeof(bf16));
  bf16*  vbuf  = (bf16*)carve((size_t)NH * DHD * SQ * sizeof(bf16));
  bf16*  obuf  = (bf16*)carve((size_t)SQ * DM * sizeof(bf16));
  bf16*  gbuf  = (bf16*)carve((size_t)SQ * FF * sizeof(bf16));

  dim3 tb(32, 8);
  transpose_cvt_kernel<<<dim3(3 * DM / 32, DM / 32, NL), tb, 0, stream>>>(
      Wqkv, wqkvT, DM, 3 * DM, (long)DM * 3 * DM, (long)3 * DM * DM);
  transpose_cvt_kernel<<<dim3(DM / 32, DM / 32, NL), tb, 0, stream>>>(
      Wo, woT, DM, DM, (long)DM * DM, (long)DM * DM);
  transpose_cvt_kernel<<<dim3(FF / 32, DM / 32, NL), tb, 0, stream>>>(
      W1, w1T, DM, FF, (long)DM * FF, (long)FF * DM);
  transpose_cvt_kernel<<<dim3(DM / 32, FF / 32, NL), tb, 0, stream>>>(
      W2, w2T, FF, DM, (long)FF * DM, (long)DM * FF);
  long nwte = (long)SV * DM;
  cvt_kernel<<<(int)((nwte + 255) / 256), 256, 0, stream>>>(wte, wteB, nwte);

  embed_kernel<<<SQ, 256, 0, stream>>>(tokens, wte, wpe, x);

  for (int l = 0; l < NL; ++l) {
    layernorm_kernel<<<SQ, 256, 0, stream>>>(x, ln1g + l * DM, ln1b + l * DM, hbuf);
    gemm_bf16_kernel<3><<<dim3(3 * DM / 128, SQ / 128), 256, 0, stream>>>(
        hbuf, wqkvT + (long)l * 3 * DM * DM, bqkv + l * 3 * DM, nullptr,
        nullptr, nullptr, qb, kbuf, vbuf, SQ, 3 * DM, DM);
    attention_kernel<<<dim3(SQ / 16, NH), 32, 0, stream>>>(qb, kbuf, vbuf, obuf);
    gemm_bf16_kernel<1><<<dim3(DM / 128, SQ / 128), 256, 0, stream>>>(
        obuf, woT + (long)l * DM * DM, bo + l * DM, x, x, nullptr,
        nullptr, nullptr, nullptr, SQ, DM, DM);
    layernorm_kernel<<<SQ, 256, 0, stream>>>(x, ln2g + l * DM, ln2b + l * DM, hbuf);
    gemm_bf16_kernel<2><<<dim3(FF / 128, SQ / 128), 256, 0, stream>>>(
        hbuf, w1T + (long)l * FF * DM, b1 + l * FF, nullptr, nullptr, gbuf,
        nullptr, nullptr, nullptr, SQ, FF, DM);
    gemm_bf16_kernel<1><<<dim3(DM / 128, SQ / 128), 256, 0, stream>>>(
        gbuf, w2T + (long)l * DM * FF, b2 + l * DM, x, x, nullptr,
        nullptr, nullptr, nullptr, SQ, DM, FF);
  }
  layernorm_kernel<<<SQ, 256, 0, stream>>>(x, lnfg, lnfb, hbuf);
  gemm_bf16_kernel<0><<<dim3((SV + 127) / 128, SQ / 128), 256, 0, stream>>>(
      hbuf, wteB, nullptr, nullptr, (float*)d_out, nullptr,
      nullptr, nullptr, nullptr, SQ, SV, DM);
}
